// SelfAttenionV2_19602230739992
// MI455X (gfx1250) — matrix operand order repaired and checked by reading, not compiled, not run on hardware
//
#include <hip/hip_runtime.h>

typedef __attribute__((ext_vector_type(16))) __bf16 v16bf;
typedef __attribute__((ext_vector_type(8)))  __bf16 v8bf;
typedef __attribute__((ext_vector_type(8)))  float  v8f;
typedef __attribute__((ext_vector_type(4)))  unsigned int u32x4;
typedef __attribute__((ext_vector_type(8)))  int i32x8;
typedef __attribute__((ext_vector_type(4)))  int i32x4;

union AV { v16bf v; v8bf h[2]; };

#define D_MODEL 1024
#define N_HEADS 16
#define D_KD    64
#define T_LEN   2048
#define MAXLEN  2048

// LDS weight tile: 64 rows x 128 elems per phase; TDM pads 32B per 256B row.
// Padded row stride = 288B = 144 elems. Per-buffer 64*288 = 18432 B, double-buffered.
#define BL_ROW_ELEMS 144
#define BL_BUF_ELEMS (64 * BL_ROW_ELEMS)
#define BL_BUF_BYTES (BL_BUF_ELEMS * 2)

// ---------------------------------------------------------------- convert
__global__ __launch_bounds__(256) void cvt_f32_bf16(const float* __restrict__ in,
                                                    __bf16* __restrict__ out, int n) {
    int i = blockIdx.x * blockDim.x + threadIdx.x;
    if (i < n) out[i] = (__bf16)in[i];
}

// TDM: DMA a 128x64 (K x N) bf16 tile of W into LDS with row padding.
__device__ __forceinline__ void tdm_load_w_tile(const __bf16* gsrc, unsigned ldsa) {
    unsigned long long ga = (unsigned long long)(size_t)gsrc;
    u32x4 g0;
    g0[0] = 1u;                                       // count=1, user mode
    g0[1] = ldsa;                                     // lds_addr (bytes)
    g0[2] = (unsigned)ga;                             // global_addr[31:0]
    g0[3] = (unsigned)((ga >> 32) & 0x1FFFFFFull)     // global_addr[56:32]
            | 0x80000000u;                            // type=2 (image)
    i32x8 g1;
    g1[0] = (int)(0x00010000u        // data_size=1 -> 2 bytes
                | 0x00100000u        // pad_enable
                | (5u << 22)         // pad_interval: 64 DW = 256B
                | (7u << 25));       // pad_amount: 8 DW = 32B
    g1[1] = (int)(128u << 16);       // tensor_dim0[15:0] = 128
    g1[2] = (int)(64u << 16);        // dim0 hi=0 | tensor_dim1[15:0] = 64
    g1[3] = (int)(128u << 16);       // dim1 hi=0 | tile_dim0 = 128
    g1[4] = 64;                      // tile_dim1 = 64, tile_dim2 = 0
    g1[5] = 1024;                    // tensor_dim0_stride = 1024 elems
    g1[6] = 0;
    g1[7] = 0;
    i32x4 g2 = {0, 0, 0, 0};
    i32x4 g3 = {0, 0, 0, 0};
    i32x8 g4 = {0, 0, 0, 0, 0, 0, 0, 0};
    __builtin_amdgcn_tensor_load_to_lds(g0, g1, g2, g3, g4, 0);
}

// ---------------------------------------------------------------- GEMM
// Y[m,n] = sum_k A[m,k] * W[n,k] + bias[n].  M=4096, N=1024, K=1024.
// Block: 8 waves, 128 rows x 64 cols. Weight tile double-buffered in LDS via
// async TDM (8 K-phases of 128); DMA of phase p+1 overlaps compute of phase p.
// mode 0: bf16 out[((b*16+head)*2048+t)*64+d]   (per-head Q/K)
// mode 1: bf16 out[((b*16+head)*64+d)*2048+t]   (transposed V)
// mode 2: f32  out[m*1024+n]                    (final output)
__global__ __launch_bounds__(256) void gemm_bf16(const __bf16* __restrict__ A,
                                                 const __bf16* __restrict__ W,
                                                 const float* __restrict__ bias,
                                                 void* __restrict__ out, int mode) {
    __shared__ __align__(128) __bf16 bl[2 * BL_BUF_ELEMS];   // 36864 B
    const int widx = threadIdx.x >> 5;
    const int lane = threadIdx.x & 31;
    const int lm = lane & 15, lh = lane >> 4;
    const int tnb = blockIdx.x & 15;          // 16 col-blocks of 64
    const int tmb = blockIdx.x >> 4;          // 32 row-blocks of 128
    const int n0 = tnb * 64;
    const int m0 = tmb * 128 + widx * 16;

    v8f acc[4] = {};
    const __bf16* arow = A + (size_t)(m0 + lm) * D_MODEL;
    const __bf16* wtile = W + (size_t)n0 * D_MODEL;   // 64 rows of W
    const unsigned ldsbase = (unsigned)(size_t)(void*)bl;

    if (widx == 0) tdm_load_w_tile(wtile, ldsbase);   // prologue: phase 0

    for (int ph = 0; ph < 8; ++ph) {
        if (widx == 0) __builtin_amdgcn_s_wait_tensorcnt(0);  // TDM(ph) done
        __syncthreads();                                      // publish buf[ph&1]
        if (widx == 0 && ph < 7)                              // async fill of other buf
            tdm_load_w_tile(wtile + (ph + 1) * 128, ldsbase + ((ph + 1) & 1) * BL_BUF_BYTES);

        const __bf16* bbuf = bl + (ph & 1) * BL_BUF_ELEMS;
        const int kbase = ph * 128;
#pragma unroll
        for (int kk = 0; kk < 128; kk += 32) {
            AV a;
            a.h[0] = *(const v8bf*)(arow + kbase + kk + 8 * lh);
            a.h[1] = *(const v8bf*)(arow + kbase + kk + 16 + 8 * lh);
            AV b[4];
#pragma unroll
            for (int j = 0; j < 4; ++j) {
                const __bf16* bp = bbuf + (j * 16 + lm) * BL_ROW_ELEMS + kk + 8 * lh;
                b[j].h[0] = *(const v8bf*)(bp);
                b[j].h[1] = *(const v8bf*)(bp + 16);
            }
#pragma unroll
            for (int j = 0; j < 4; ++j)
                acc[j] = __builtin_amdgcn_wmma_f32_16x16x32_bf16(
                    false, a.v, false, b[j].v, (short)0, acc[j], false, false);
        }
        __syncthreads();   // buf[ph&1] readers done before TDM(ph+2) overwrites it
    }

    if (mode == 2) {
        float* o = (float*)out;
#pragma unroll
        for (int j = 0; j < 4; ++j)
#pragma unroll
            for (int r = 0; r < 8; ++r) {
                int m = m0 + r + 8 * lh;
                int n = n0 + j * 16 + lm;
                o[(size_t)m * D_MODEL + n] = acc[j][r] + bias[n];
            }
    } else if (mode == 0) {
        __bf16* q = (__bf16*)out;
        const int head = tnb;
#pragma unroll
        for (int j = 0; j < 4; ++j)
#pragma unroll
            for (int r = 0; r < 8; ++r) {
                int m = m0 + r + 8 * lh;
                int b = m >> 11, t = m & (T_LEN - 1);
                int d = j * 16 + lm;
                float v = acc[j][r] + bias[n0 + d];
                q[(((size_t)(b * N_HEADS + head)) * T_LEN + t) * D_KD + d] = (__bf16)v;
            }
    } else {                                  // mode 1: transposed V
        __bf16* vt = (__bf16*)out;
        const int head = tnb;
#pragma unroll
        for (int j = 0; j < 4; ++j)
#pragma unroll
            for (int r = 0; r < 8; ++r) {
                int m = m0 + r + 8 * lh;
                int b = m >> 11, t = m & (T_LEN - 1);
                int d = j * 16 + lm;
                float v = acc[j][r] + bias[n0 + d];
                vt[((size_t)(b * N_HEADS + head) * D_KD + d) * T_LEN + t] = (__bf16)v;
            }
    }
}

// ---------------------------------------------------------------- attention
// One wave per (bh, 16-query tile). Flash-style online softmax over 32-key pairs.
__global__ __launch_bounds__(256) void attn_kernel(const __bf16* __restrict__ Qb,
                                                   const __bf16* __restrict__ Kb,
                                                   const __bf16* __restrict__ Vt,
                                                   const float* __restrict__ table,
                                                   __bf16* __restrict__ attn_flat) {
    __shared__ __align__(16) unsigned short plds_raw[8][16 * 32];
    const int wave = (blockIdx.x * blockDim.x + threadIdx.x) >> 5;
    const int widx = threadIdx.x >> 5;
    const int lane = threadIdx.x & 31;
    const int lm = lane & 15, lh = lane >> 4;
    const int qt = wave & 127;         // 128 query tiles
    const int bh = wave >> 7;          // 32 (b,h) pairs
    const int head = bh & 15, b = bh >> 4;

    const __bf16* qrow = Qb + ((size_t)bh * T_LEN + qt * 16 + lm) * D_KD;
    AV aq0, aq1;
    aq0.h[0] = *(const v8bf*)(qrow + 16 * lh);
    aq0.h[1] = *(const v8bf*)(qrow + 8 + 16 * lh);
    aq1.h[0] = *(const v8bf*)(qrow + 32 + 16 * lh);
    aq1.h[1] = *(const v8bf*)(qrow + 40 + 16 * lh);

    float mrow[8], lrow[8];
#pragma unroll
    for (int r = 0; r < 8; ++r) { mrow[r] = -1e30f; lrow[r] = 0.f; }
    v8f o[4] = {};

    int tq[8];
#pragma unroll
    for (int r = 0; r < 8; ++r) tq[r] = qt * 16 + r + 8 * lh;

    __bf16* pl = (__bf16*)plds_raw[widx];
    const int ktmax = (qt * 16 + 15) >> 5;

    for (int kt2 = 0; kt2 <= ktmax; ++kt2) {
        const int kb = kt2 * 32;
        v8f s[2] = {};
#pragma unroll
        for (int t01 = 0; t01 < 2; ++t01) {
            const __bf16* krow = Kb + ((size_t)bh * T_LEN + kb + t01 * 16 + lm) * D_KD;
            v16bf b0 = *(const v16bf*)(krow + 16 * lh);
            v16bf b1 = *(const v16bf*)(krow + 32 + 16 * lh);
            s[t01] = __builtin_amdgcn_wmma_f32_16x16x32_bf16(
                false, aq0.v, false, b0, (short)0, s[t01], false, false);
            s[t01] = __builtin_amdgcn_wmma_f32_16x16x32_bf16(
                false, aq1.v, false, b1, (short)0, s[t01], false, false);
        }
        // scale + rel-pos bias + causal mask
#pragma unroll
        for (int t01 = 0; t01 < 2; ++t01) {
            int tk = kb + t01 * 16 + lm;
#pragma unroll
            for (int r = 0; r < 8; ++r) {
                float v = s[t01][r] * 0.125f;   // 1/sqrt(64)
                v += table[(size_t)(tq[r] - tk + (MAXLEN - 1)) * N_HEADS + head];
                s[t01][r] = (tk > tq[r]) ? -1e30f : v;
            }
        }
        // online softmax
        float alpha[8];
#pragma unroll
        for (int r = 0; r < 8; ++r) {
            float mx = fmaxf(s[0][r], s[1][r]);
            mx = fmaxf(mx, __shfl_xor(mx, 1, 32));
            mx = fmaxf(mx, __shfl_xor(mx, 2, 32));
            mx = fmaxf(mx, __shfl_xor(mx, 4, 32));
            mx = fmaxf(mx, __shfl_xor(mx, 8, 32));
            float mn = fmaxf(mrow[r], mx);
            alpha[r] = __expf(mrow[r] - mn);
            float p0 = __expf(s[0][r] - mn);
            float p1 = __expf(s[1][r] - mn);
            s[0][r] = p0; s[1][r] = p1;
            float ts = p0 + p1;
            ts += __shfl_xor(ts, 1, 32);
            ts += __shfl_xor(ts, 2, 32);
            ts += __shfl_xor(ts, 4, 32);
            ts += __shfl_xor(ts, 8, 32);
            lrow[r] = lrow[r] * alpha[r] + ts;
            mrow[r] = mn;
        }
#pragma unroll
        for (int j = 0; j < 4; ++j)
#pragma unroll
            for (int r = 0; r < 8; ++r) o[j][r] *= alpha[r];

        // P (C-layout) -> LDS row-major 16x32 -> A-layout fragments
#pragma unroll
        for (int r = 0; r < 8; ++r) {
            int m = r + 8 * lh;
            pl[m * 32 + lm]      = (__bf16)s[0][r];
            pl[m * 32 + 16 + lm] = (__bf16)s[1][r];
        }
        asm volatile("s_wait_dscnt 0x0" ::: "memory");
        AV ap;
        ap.h[0] = *(const v8bf*)(pl + lm * 32 + 16 * lh);
        ap.h[1] = *(const v8bf*)(pl + lm * 32 + 8 + 16 * lh);
        AV bv[4];
#pragma unroll
        for (int j = 0; j < 4; ++j) {
            const __bf16* vp = Vt + ((size_t)bh * D_KD + j * 16 + lm) * T_LEN + kb + 16 * lh;
            bv[j].h[0] = *(const v8bf*)(vp);
            bv[j].h[1] = *(const v8bf*)(vp + 8);
        }
#pragma unroll
        for (int j = 0; j < 4; ++j)
            o[j] = __builtin_amdgcn_wmma_f32_16x16x32_bf16(
                false, ap.v, false, bv[j].v, (short)0, o[j], false, false);
    }

    // normalize + store [b, t, head*64+d] bf16
#pragma unroll
    for (int j = 0; j < 4; ++j)
#pragma unroll
        for (int r = 0; r < 8; ++r) {
            float v = o[j][r] / lrow[r];
            attn_flat[((size_t)b * T_LEN + tq[r]) * D_MODEL + head * D_KD + j * 16 + lm] = (__bf16)v;
        }
}

// ---------------------------------------------------------------- launch
extern "C" void kernel_launch(void* const* d_in, const int* in_sizes, int n_in,
                              void* d_out, int out_size, void* d_ws, size_t ws_size,
                              hipStream_t stream) {
    (void)in_sizes; (void)n_in; (void)out_size; (void)ws_size;
    const float* x  = (const float*)d_in[0];
    const float* WQ = (const float*)d_in[1];
    const float* bQ = (const float*)d_in[2];
    const float* WK = (const float*)d_in[3];
    const float* bK = (const float*)d_in[4];
    const float* WV = (const float*)d_in[5];
    const float* bV = (const float*)d_in[6];
    const float* WO = (const float*)d_in[7];
    const float* bO = (const float*)d_in[8];
    const float* table = (const float*)d_in[9];

    size_t off = 0;
    char* base = (char*)d_ws;
    auto take = [&](size_t bytes) { char* p = base + off; off += bytes; return p; };
    const size_t SZ_X = (size_t)4096 * 1024 * 2;   // 8 MB bf16
    const size_t SZ_W = (size_t)1024 * 1024 * 2;   // 2 MB bf16
    __bf16* xb    = (__bf16*)take(SZ_X);
    __bf16* wqb   = (__bf16*)take(SZ_W);
    __bf16* wkb   = (__bf16*)take(SZ_W);
    __bf16* wvb   = (__bf16*)take(SZ_W);
    __bf16* wob   = (__bf16*)take(SZ_W);
    __bf16* Qb    = (__bf16*)take(SZ_X);
    __bf16* Kb    = (__bf16*)take(SZ_X);
    __bf16* Vt    = (__bf16*)take(SZ_X);
    __bf16* attnf = (__bf16*)take(SZ_X);

    const int nx = 4096 * 1024, nw = 1024 * 1024;
    cvt_f32_bf16<<<(nx + 255) / 256, 256, 0, stream>>>(x, xb, nx);
    cvt_f32_bf16<<<(nw + 255) / 256, 256, 0, stream>>>(WQ, wqb, nw);
    cvt_f32_bf16<<<(nw + 255) / 256, 256, 0, stream>>>(WK, wkb, nw);
    cvt_f32_bf16<<<(nw + 255) / 256, 256, 0, stream>>>(WV, wvb, nw);
    cvt_f32_bf16<<<(nw + 255) / 256, 256, 0, stream>>>(WO, wob, nw);

    // 512 blocks x 8 waves: each block = 128 rows x 64 cols
    gemm_bf16<<<512, 256, 0, stream>>>(xb, wqb, bQ, Qb, 0);
    gemm_bf16<<<512, 256, 0, stream>>>(xb, wkb, bK, Kb, 0);
    gemm_bf16<<<512, 256, 0, stream>>>(xb, wvb, bV, Vt, 1);

    attn_kernel<<<512, 256, 0, stream>>>(Qb, Kb, Vt, table, attnf);

    gemm_bf16<<<512, 256, 0, stream>>>(attnf, wob, bO, d_out, 2);
}
